// CVQVAE_33603824124064
// MI455X (gfx1250) — compile-verified, matrix operand order checked by reading, not run on hardware
//
#include <hip/hip_runtime.h>
#include <hip/hip_bf16.h>

typedef __attribute__((ext_vector_type(16))) __bf16 v16bf;
typedef __attribute__((ext_vector_type(8)))  float  v8f;

#define LATENT 128
#define KCODE  1024
#define LEN    128
#define HID    200
#define HIDP   208   // 13*16  (N-pad of 200)
#define HPK    224   // 7*32   (K-pad of 200)
#define INF_   768
#define CONDW  1536
#define BATCH  256
#define G4     800   // 4*HID
#define D1IN   2432  // LATENT + CONDW + INF_
#define H2W    416   // 13*32  (pad of 400)
#define MROWS  32768 // BATCH*LEN

// ---------------------------------------------------------------------------
// CDNA5 16-bit A/B fragment layout (ISA 7.12.2): lane L holds row L%16,
// half = L/16; slot s -> k = s + 8*half + (s>=8 ? 8 : 0), i.e. two contiguous
// 8-element groups at k+8*half and k+16+8*half.
// ---------------------------------------------------------------------------
__device__ __forceinline__ v16bf frag_from_bf16(const __bf16* __restrict__ row,
                                                int half) {
  const uint4* p0 = (const uint4*)(row + 8 * half);
  const uint4* p1 = (const uint4*)(row + 16 + 8 * half);
  union { uint4 u[2]; v16bf v; } w;
  w.u[0] = *p0;
  w.u[1] = *p1;
  return w.v;
}

__device__ __forceinline__ v16bf frag_from_f32(const float* __restrict__ row,
                                               int k0, int half) {
  const float* pa = row + k0 + 8 * half;
  float4 a0 = *(const float4*)(pa);
  float4 a1 = *(const float4*)(pa + 4);
  float4 b0 = *(const float4*)(pa + 16);
  float4 b1 = *(const float4*)(pa + 20);
  v16bf v;
  v[0]  = (__bf16)a0.x; v[1]  = (__bf16)a0.y; v[2]  = (__bf16)a0.z; v[3]  = (__bf16)a0.w;
  v[4]  = (__bf16)a1.x; v[5]  = (__bf16)a1.y; v[6]  = (__bf16)a1.z; v[7]  = (__bf16)a1.w;
  v[8]  = (__bf16)b0.x; v[9]  = (__bf16)b0.y; v[10] = (__bf16)b0.z; v[11] = (__bf16)b0.w;
  v[12] = (__bf16)b1.x; v[13] = (__bf16)b1.y; v[14] = (__bf16)b1.z; v[15] = (__bf16)b1.w;
  return v;
}

__device__ __forceinline__ v16bf frag_from_lds(const __bf16* tileA, int lm, int half) {
  union { uint4 u[2]; v16bf v; } w;
  w.u[0] = *(const uint4*)(tileA + lm * 32 + 8 * half);
  w.u[1] = *(const uint4*)(tileA + lm * 32 + 16 + 8 * half);
  return w.v;
}

__device__ __forceinline__ float sigmoidf_(float x) {
  return 1.0f / (1.0f + __expf(-x));
}

// ---------------------------------------------------------------------------
// Weight conversion f32 -> padded bf16.  dst layout: [groups][dstR][dstC].
// ---------------------------------------------------------------------------
__global__ void convert_pad_kernel(const float* __restrict__ src, __bf16* __restrict__ dst,
                                   int groups, int srcR, int dstR, int srcC, int dstC,
                                   int total) {
  int i = blockIdx.x * blockDim.x + threadIdx.x;
  if (i >= total) return;
  int perG = dstR * dstC;
  int g = i / perG;
  int rem = i - g * perG;
  int r = rem / dstC;
  int c = rem - r * dstC;
  float v = (r < srcR && c < srcC) ? src[((size_t)g * srcR + r) * srcC + c] : 0.0f;
  dst[i] = (__bf16)v;
}

// Zero h/c ping-pong buffers, build combined bias (b_ih + b_hh), zero barrier.
__global__ void init_kernel(float* __restrict__ hbuf, float* __restrict__ cbuf,
                            float* __restrict__ bias800, int* __restrict__ bar,
                            const float* __restrict__ b_ih, const float* __restrict__ b_hh) {
  int i = blockIdx.x * blockDim.x + threadIdx.x;
  const int HN = 2 * BATCH * HPK;
  const int CN = 2 * BATCH * HIDP;
  if (i < HN) hbuf[i] = 0.0f;
  else if (i < HN + CN) cbuf[i - HN] = 0.0f;
  else if (i < HN + CN + G4) { int j = i - HN - CN; bias800[j] = b_ih[j] + b_hh[j]; }
  else if (i == HN + CN + G4) *bar = 0;
}

// ---------------------------------------------------------------------------
// Generic tiled GEMM:  C = act( A * B^T + bias )
//   A: [M x K], f32 or bf16 (ABF16), staged per 16x32 chunk into LDS (bf16)
//      and shared by all 4 waves of the block.
//   B: [N][K] bf16 (row = output column), K padded to x32, rows >= nreal zero.
//   Block = 128 threads = 4 waves; each wave owns 4 N-tiles.  Out-of-range
//   tiles are CLAMPED to a valid B row (branch-free inner loop, no spills);
//   validity is applied only at store time.
//   ACT: 0=none, 1=relu, 2=sigmoid.  CBF16: store bf16 instead of f32.
// ---------------------------------------------------------------------------
template <bool ABF16, int ACT, bool CBF16>
__global__ __launch_bounds__(128, 1) void gemm_wmma(
    const void* __restrict__ Av, int lda,
    const __bf16* __restrict__ Bm, int ldb,
    const float* __restrict__ bias, int nreal,
    void* __restrict__ Cv, int ldc,
    int K, int ntiles) {
  __shared__ __align__(16) __bf16 tileA[16 * 32];
  const int tid  = threadIdx.x;
  const int lane = tid & 31;
  const int wave = tid >> 5;
  const int m0   = blockIdx.x * 16;
  const int half = lane >> 4;
  const int lm   = lane & 15;

  v8f acc[4];
  const __bf16* brow[4];
  bool valid[4];
  int ncol[4];
#pragma unroll
  for (int j = 0; j < 4; ++j) {
    int nt   = blockIdx.y * 16 + wave * 4 + j;
    valid[j] = nt < ntiles;
    int ntc  = valid[j] ? nt : (ntiles - 1);      // clamp to a valid B row
    ncol[j]  = ntc * 16 + lm;
    brow[j]  = Bm + (size_t)ncol[j] * ldb;
    float bv = (valid[j] && ncol[j] < nreal) ? bias[ncol[j]] : 0.0f;
#pragma unroll
    for (int i = 0; i < 8; ++i) acc[j][i] = bv;
  }

  // cooperative staging: thread -> 4 consecutive elements of the 16x32 chunk
  const int srow = tid >> 3;        // 0..15
  const int scol = (tid & 7) * 4;   // 0,4,..,28

  for (int k0 = 0; k0 < K; k0 += 32) {
    __syncthreads();
    if (ABF16) {
      const __bf16* src = (const __bf16*)Av + (size_t)(m0 + srow) * lda + k0 + scol;
      *(uint2*)(tileA + srow * 32 + scol) = *(const uint2*)src;
    } else {
      const float* src = (const float*)Av + (size_t)(m0 + srow) * lda + k0 + scol;
      float4 f = *(const float4*)src;
      union { __bf16 h[4]; uint2 u; } pk;
      pk.h[0] = (__bf16)f.x; pk.h[1] = (__bf16)f.y;
      pk.h[2] = (__bf16)f.z; pk.h[3] = (__bf16)f.w;
      *(uint2*)(tileA + srow * 32 + scol) = pk.u;
    }
    __syncthreads();

    v16bf af = frag_from_lds(tileA, lm, half);
#pragma unroll
    for (int j = 0; j < 4; ++j) {
      v16bf bf = frag_from_bf16(brow[j] + k0, half);
      acc[j] = __builtin_amdgcn_wmma_f32_16x16x32_bf16(false, af, false, bf,
                                                       (short)0, acc[j], false, false);
    }
  }

#pragma unroll
  for (int j = 0; j < 4; ++j) {
    if (!valid[j]) continue;
#pragma unroll
    for (int r = 0; r < 8; ++r) {
      int m = m0 + r + 8 * half;
      float v = acc[j][r];
      if (ACT == 1)      v = v > 0.0f ? v : 0.0f;
      else if (ACT == 2) v = sigmoidf_(v);
      if (CBF16) ((__bf16*)Cv)[(size_t)m * ldc + ncol[j]] = (__bf16)v;
      else       ((float*)Cv)[(size_t)m * ldc + ncol[j]]  = v;
    }
  }
}

// ---------------------------------------------------------------------------
// Decoder layer-1 GEMM with fused concat A = [z_q | condition | noise].
// The K loop is split into the three source regions (boundaries 128/1664 are
// 32-aligned) so the inner loops are branch-free.  Same 4-wave / 4-tiles /
// LDS-staged-A structure; ntiles = 14 (N pad 200 -> 224); output bf16.
// ---------------------------------------------------------------------------
#define DEC1_CHUNK(SRCPTR, KK, KB)                                              \
  {                                                                             \
    __syncthreads();                                                            \
    float4 f = *(const float4*)((SRCPTR) + (KK) + scol);                        \
    union { __bf16 h[4]; uint2 u; } pk;                                         \
    pk.h[0] = (__bf16)f.x; pk.h[1] = (__bf16)f.y;                               \
    pk.h[2] = (__bf16)f.z; pk.h[3] = (__bf16)f.w;                               \
    *(uint2*)(tileA + srow * 32 + scol) = pk.u;                                 \
    __syncthreads();                                                            \
    v16bf af = frag_from_lds(tileA, lm, half);                                  \
    _Pragma("unroll")                                                           \
    for (int j = 0; j < 4; ++j) {                                               \
      v16bf bf = frag_from_bf16(brow[j] + (KB), half);                          \
      acc[j] = __builtin_amdgcn_wmma_f32_16x16x32_bf16(false, af, false, bf,    \
                                                       (short)0, acc[j], false, false); \
    }                                                                           \
  }

__global__ __launch_bounds__(128, 1) void wmma_dec1(
    const float* __restrict__ zq, const float* __restrict__ cond,
    const float* __restrict__ noise, const __bf16* __restrict__ W1b,
    const float* __restrict__ b1, __bf16* __restrict__ h1) {
  __shared__ __align__(16) __bf16 tileA[16 * 32];
  const int tid  = threadIdx.x;
  const int lane = tid & 31;
  const int wave = tid >> 5;
  const int m0   = blockIdx.x * 16;
  const int half = lane >> 4;
  const int lm   = lane & 15;
  const int NT   = HPK / 16;   // 14

  v8f acc[4];
  const __bf16* brow[4];
  bool valid[4];
  int ncol[4];
#pragma unroll
  for (int j = 0; j < 4; ++j) {
    int nt   = wave * 4 + j;
    valid[j] = nt < NT;
    int ntc  = valid[j] ? nt : (NT - 1);
    ncol[j]  = ntc * 16 + lm;
    brow[j]  = W1b + (size_t)ncol[j] * D1IN;
    float bv = (valid[j] && ncol[j] < HID) ? b1[ncol[j]] : 0.0f;
#pragma unroll
    for (int i = 0; i < 8; ++i) acc[j][i] = bv;
  }

  const int srow = tid >> 3;
  const int scol = (tid & 7) * 4;
  const int ms   = m0 + srow;
  const int bbs  = ms >> 7;       // batch index of staging row
  const float* zr = zq    + (size_t)bbs * LATENT;
  const float* cr = cond  + (size_t)ms * CONDW;
  const float* nr = noise + (size_t)bbs * INF_;

  for (int k0 = 0; k0 < LATENT; k0 += 32) DEC1_CHUNK(zr, k0, k0);
  for (int k0 = 0; k0 < CONDW; k0 += 32)  DEC1_CHUNK(cr, k0, LATENT + k0);
  for (int k0 = 0; k0 < INF_; k0 += 32)   DEC1_CHUNK(nr, k0, LATENT + CONDW + k0);

#pragma unroll
  for (int j = 0; j < 4; ++j) {
    if (!valid[j]) continue;
#pragma unroll
    for (int r = 0; r < 8; ++r) {
      int mm = m0 + r + 8 * half;
      float v = acc[j][r];
      v = v > 0.0f ? v : 0.0f;
      h1[(size_t)mm * HPK + ncol[j]] = (__bf16)v;
    }
  }
}

// ---------------------------------------------------------------------------
// Persistent LSTM scan: 16x13 = 208 single-wave blocks, one (batch16 x hid16)
// tile each.  Loop-interchanged: per K-chunk one A fragment feeds 4 gate
// WMMAs (minimal live registers).  Grid-wide monotonic atomic barrier/step.
// ---------------------------------------------------------------------------
__global__ __launch_bounds__(32, 1) void lstm_scan(
    const float* __restrict__ Gx, const __bf16* __restrict__ Whhb,
    float* __restrict__ hbuf, float* __restrict__ cbuf,
    int* __restrict__ bar) {
  const int nblk = 16 * 13;
  int lane = threadIdx.x & 31;
  int half = lane >> 4;
  int lm = lane & 15;
  int m0 = blockIdx.x * 16;              // batch tile
  int n0 = blockIdx.y * 16;              // hidden tile
  int ncol = n0 + lm;

  const __bf16* brow[4];
#pragma unroll
  for (int g = 0; g < 4; ++g)
    brow[g] = Whhb + (size_t)(g * HIDP + n0 + lm) * HPK;

  for (int t = 0; t < LEN; ++t) {
    const float* hprev = hbuf + (size_t)(t & 1) * (BATCH * HPK);
    float*       hnext = hbuf + (size_t)((t + 1) & 1) * (BATCH * HPK);
    const float* cprev = cbuf + (size_t)(t & 1) * (BATCH * HIDP);
    float*       cnext = cbuf + (size_t)((t + 1) & 1) * (BATCH * HIDP);

    const float* arow = hprev + (size_t)(m0 + lm) * HPK;
    v8f acc[4];
#pragma unroll
    for (int g = 0; g < 4; ++g)
#pragma unroll
      for (int i = 0; i < 8; ++i) acc[g][i] = 0.0f;

#pragma unroll
    for (int kc = 0; kc < HPK / 32; ++kc) {
      v16bf af = frag_from_f32(arow, kc * 32, half);
#pragma unroll
      for (int g = 0; g < 4; ++g) {
        v16bf bf = frag_from_bf16(brow[g] + kc * 32, half);
        acc[g] = __builtin_amdgcn_wmma_f32_16x16x32_bf16(false, af, false, bf,
                                                         (short)0, acc[g], false, false);
      }
    }

    if (ncol < HID) {
#pragma unroll
      for (int r = 0; r < 8; ++r) {
        int b = m0 + r + 8 * half;
        const float* gx = Gx + ((size_t)b * LEN + t) * G4;
        float iv = acc[0][r] + gx[ncol];
        float fv = acc[1][r] + gx[HID + ncol];
        float gv = acc[2][r] + gx[2 * HID + ncol];
        float ov = acc[3][r] + gx[3 * HID + ncol];
        float cp = cprev[(size_t)b * HIDP + ncol];
        float cn = sigmoidf_(fv) * cp + sigmoidf_(iv) * tanhf(gv);
        float hn = sigmoidf_(ov) * tanhf(cn);
        cnext[(size_t)b * HIDP + ncol] = cn;
        hnext[(size_t)b * HPK + ncol] = hn;
      }
    }

    // grid-wide barrier (monotonic counter, no reset)
    __threadfence();
    if (threadIdx.x == 0) {
      __hip_atomic_fetch_add(bar, 1, __ATOMIC_ACQ_REL, __HIP_MEMORY_SCOPE_AGENT);
      int target = nblk * (t + 1);
      while (__hip_atomic_load(bar, __ATOMIC_ACQUIRE, __HIP_MEMORY_SCOPE_AGENT) < target)
        __builtin_amdgcn_s_sleep(1);
    }
    __syncthreads();
    __threadfence();
  }
}

// ---------------------------------------------------------------------------
// Vector-quantizer: per-batch argmin of squared distance, gather codebook row.
// f32 VALU on purpose (argmin selection precision).
// ---------------------------------------------------------------------------
__global__ void vq_kernel(const float* __restrict__ ze, const float* __restrict__ emb,
                          float* __restrict__ zq) {
  __shared__ float zs[LATENT];
  __shared__ float sd[128];
  __shared__ int   si[128];
  int b = blockIdx.x;
  int tid = threadIdx.x;
  zs[tid] = ze[(size_t)b * LATENT + tid];  // blockDim == 128 == LATENT
  __syncthreads();

  float best = 3.4e38f;
  int bi = 0x7fffffff;
  for (int j = tid; j < KCODE; j += 128) {
    const float* e = emb + (size_t)j * LATENT;
    float d = 0.0f;
#pragma unroll 4
    for (int k = 0; k < LATENT; ++k) {
      float df = zs[k] - e[k];
      d += df * df;
    }
    if (d < best || (d == best && j < bi)) { best = d; bi = j; }
  }
  sd[tid] = best;
  si[tid] = bi;
  __syncthreads();
  for (int s = 64; s > 0; s >>= 1) {
    if (tid < s) {
      if (sd[tid + s] < sd[tid] || (sd[tid + s] == sd[tid] && si[tid + s] < si[tid])) {
        sd[tid] = sd[tid + s];
        si[tid] = si[tid + s];
      }
    }
    __syncthreads();
  }
  int idx = si[0];
  zq[(size_t)b * LATENT + tid] = emb[(size_t)idx * LATENT + tid];
}

// ---------------------------------------------------------------------------
extern "C" void kernel_launch(void* const* d_in, const int* in_sizes, int n_in,
                              void* d_out, int out_size, void* d_ws, size_t ws_size,
                              hipStream_t stream) {
  (void)in_sizes; (void)n_in; (void)out_size; (void)ws_size;
  const float* x     = (const float*)d_in[0];
  const float* cond  = (const float*)d_in[1];
  const float* noise = (const float*)d_in[2];
  const float* W_ih  = (const float*)d_in[3];
  const float* W_hh  = (const float*)d_in[4];
  const float* b_ih  = (const float*)d_in[5];
  const float* b_hh  = (const float*)d_in[6];
  const float* W_enc = (const float*)d_in[7];
  const float* b_enc = (const float*)d_in[8];
  const float* emb   = (const float*)d_in[9];
  const float* W1    = (const float*)d_in[10];
  const float* b1    = (const float*)d_in[11];
  const float* W2    = (const float*)d_in[12];
  const float* b2    = (const float*)d_in[13];
  const float* W3    = (const float*)d_in[14];
  const float* b3    = (const float*)d_in[15];
  float* out = (float*)d_out;

  // workspace carve-up (256B aligned)
  char* p = (char*)d_ws;
  auto carve = [&](size_t bytes) -> char* {
    char* r = p;
    p += (bytes + 255) & ~(size_t)255;
    return r;
  };
  float*  Gx      = (float*)carve((size_t)MROWS * G4 * 4);     // 104.9 MB f32
  __bf16* h1      = (__bf16*)carve((size_t)MROWS * HPK * 2);   //  14.7 MB bf16
  __bf16* h2      = (__bf16*)carve((size_t)MROWS * H2W * 2);   //  27.3 MB bf16
  float*  hbuf    = (float*)carve((size_t)2 * BATCH * HPK * 4);
  float*  cbuf    = (float*)carve((size_t)2 * BATCH * HIDP * 4);
  float*  ze      = (float*)carve((size_t)BATCH * LATENT * 4);
  float*  zq      = (float*)carve((size_t)BATCH * LATENT * 4);
  float*  bias800 = (float*)carve(G4 * 4);
  int*    bar     = (int*)carve(256);
  __bf16* Wihb    = (__bf16*)carve((size_t)G4 * INF_ * 2);
  __bf16* Whhb    = (__bf16*)carve((size_t)4 * HIDP * HPK * 2);
  __bf16* Wencb   = (__bf16*)carve((size_t)LATENT * HPK * 2);
  __bf16* W1b     = (__bf16*)carve((size_t)HPK * D1IN * 2);
  __bf16* W2b     = (__bf16*)carve((size_t)H2W * HPK * 2);
  __bf16* W3b     = (__bf16*)carve((size_t)INF_ * H2W * 2);

  auto cvt = [&](const float* src, __bf16* dst, int groups, int srcR, int dstR,
                 int srcC, int dstC) {
    int total = groups * dstR * dstC;
    convert_pad_kernel<<<(total + 255) / 256, 256, 0, stream>>>(
        src, dst, groups, srcR, dstR, srcC, dstC, total);
  };
  cvt(W_ih,  Wihb,  1, G4,     G4,     INF_, INF_);
  cvt(W_hh,  Whhb,  4, HID,    HIDP,   HID,  HPK);
  cvt(W_enc, Wencb, 1, LATENT, LATENT, HID,  HPK);
  cvt(W1,    W1b,   1, HID,    HPK,    D1IN, D1IN);
  cvt(W2,    W2b,   1, 400,    H2W,    HID,  HPK);
  cvt(W3,    W3b,   1, INF_,   INF_,   400,  H2W);

  {
    int total = 2 * BATCH * HPK + 2 * BATCH * HIDP + G4 + 1;
    init_kernel<<<(total + 255) / 256, 256, 0, stream>>>(hbuf, cbuf, bias800, bar,
                                                         b_ih, b_hh);
  }

  dim3 blk(128);
  // gates_x = x @ W_ih^T + (b_ih + b_hh)   -> Gx [32768 x 800] f32
  gemm_wmma<false, 0, false><<<dim3(MROWS / 16, 4), blk, 0, stream>>>(
      x, INF_, Wihb, INF_, bias800, G4, Gx, G4, INF_, 50);

  // LSTM scan (persistent; final h lands in hbuf slot 0 after 128 steps)
  lstm_scan<<<dim3(16, 13), 32, 0, stream>>>(Gx, Whhb, hbuf, cbuf, bar);

  // z_e = h @ W_enc^T + b_enc   -> ze [256 x 128] f32
  gemm_wmma<false, 0, false><<<dim3(BATCH / 16, 1), blk, 0, stream>>>(
      hbuf, HPK, Wencb, HPK, b_enc, LATENT, ze, LATENT, HPK, LATENT / 16);

  // vector quantize -> zq [256 x 128] f32
  vq_kernel<<<BATCH, 128, 0, stream>>>(ze, emb, zq);

  // decoder layer 1 (fused concat) -> h1 [32768 x 224] bf16
  wmma_dec1<<<dim3(MROWS / 16, 1), blk, 0, stream>>>(zq, cond, noise, W1b, b1, h1);

  // decoder layer 2 -> h2 [32768 x 416] bf16
  gemm_wmma<true, 1, true><<<dim3(MROWS / 16, 2), blk, 0, stream>>>(
      h1, HPK, W2b, HPK, b2, 400, h2, H2W, HPK, 26);

  // decoder layer 3 + sigmoid -> out [32768 x 768] f32
  gemm_wmma<true, 2, false><<<dim3(MROWS / 16, 3), blk, 0, stream>>>(
      h2, H2W, W3b, H2W, b3, INF_, out, INF_, H2W, 48);
}
